// EncoderBlock_51496657879664
// MI455X (gfx1250) — matrix operand order repaired and checked by reading, not compiled, not run on hardware
//
#include <hip/hip_runtime.h>
#include <hip/hip_bf16.h>
#include <cstdint>

// ---------------------------------------------------------------------------
// EncoderBlock for MI455X (gfx1250, wave32, WMMA 16x16x32 f16->f32)
// B=32 S=512 F=256 K=64 NH=8 D=32 CK=5 H=128
// ---------------------------------------------------------------------------

typedef __attribute__((ext_vector_type(16))) _Float16 v16h;
typedef __attribute__((ext_vector_type(8)))  float    v8f;

#define BB 32
#define SS 512
#define FF 256
#define HH 128
#define NTOK (BB * SS)          // 16384
#define NHH 8
#define DD 32
#define CKK 5
#define SLOPE 0.1f
#define EPSV 1e-5f

__device__ __forceinline__ v8f wmma16(v16h a, v16h b, v8f c) {
  // D = A(16x32,f16) x B(32x16,f16) + C(16x16,f32)
  return __builtin_amdgcn_wmma_f32_16x16x32_f16(false, a, false, b, (short)0, c,
                                                false, false);
}

// ---- fragment loaders (contiguous k order k = 16*(lane/16) + i for A and B, wave32) ---------------
// A 16x32 f16 tile stored row-major [16][ld] in LDS.
// lane m (<16): K 0..15 ; lane m+16: K 16..31
__device__ __forceinline__ v16h ld_a_frag(const _Float16* base, int ld) {
  const int lane = threadIdx.x & 31;
  const int m = lane & 15;
  const int kh = (lane >> 4) << 4;   // 0 or 16
  const _Float16* p = base + m * ld;
  v16h f;
#pragma unroll
  for (int i = 0; i < 16; ++i) f[i] = p[kh + i];
  return f;
}

// B 32x16 f16 tile stored K-contiguous per N-row: Bt[n][k], [16][ld] in LDS.
// lane n (<16): K 0..15 ; lane n+16: K 16..31
__device__ __forceinline__ v16h ld_b_frag(const _Float16* base, int ld) {
  const int lane = threadIdx.x & 31;
  const int n = lane & 15;
  const int ko = (lane >> 4) << 4;   // 0 or 16
  const _Float16* p = base + n * ld + ko;
  v16h f;
#pragma unroll
  for (int i = 0; i < 16; ++i) f[i] = p[i];
  return f;
}

// ---------------------------------------------------------------------------
// Generic WMMA GEMM:  C[M,N] = A[M,K] * W[N,K]^T (+bias)
// block = 256 thr (8 waves); tile 64x64; wave does one 16x32 C strip.
// All M,N,K are multiples of 64/32 here.
// ---------------------------------------------------------------------------
template <int OUT_HALF>
__global__ void wmma_gemm_nk(const float* __restrict__ A,
                             const float* __restrict__ Wnk,
                             const float* __restrict__ bias,
                             void* __restrict__ Cout,
                             int M, int N, int Kd) {
  __shared__ _Float16 As[64 * 32];
  __shared__ _Float16 Bs[64 * 32];
  const int tid  = threadIdx.x;
  const int wave = tid >> 5;
  const int m0 = blockIdx.x * 64;
  const int n0 = blockIdx.y * 64;
  const int wm = (wave & 3) * 16;
  const int wn = (wave >> 2) * 32;
  v8f acc0 = {}; v8f acc1 = {};

  for (int k0 = 0; k0 < Kd; k0 += 32) {
    for (int i = tid; i < 64 * 32; i += 256) {
      const int r = i >> 5, c = i & 31;
      As[i] = (_Float16)A[(size_t)(m0 + r) * Kd + k0 + c];
    }
    for (int i = tid; i < 64 * 32; i += 256) {
      const int r = i >> 5, c = i & 31;
      Bs[i] = (_Float16)Wnk[(size_t)(n0 + r) * Kd + k0 + c];
    }
    __syncthreads();
    const v16h af  = ld_a_frag(&As[wm * 32], 32);
    const v16h bf0 = ld_b_frag(&Bs[wn * 32], 32);
    const v16h bf1 = ld_b_frag(&Bs[(wn + 16) * 32], 32);
    acc0 = wmma16(af, bf0, acc0);
    acc1 = wmma16(af, bf1, acc1);
    __syncthreads();
  }

  const int lane = tid & 31;
  const int n  = lane & 15;
  const int mb = (lane >> 4) * 8;
#pragma unroll
  for (int t = 0; t < 2; ++t) {
    const v8f acc = t ? acc1 : acc0;
    const int col = n0 + wn + t * 16 + n;
    const float bv = bias ? bias[col] : 0.0f;
#pragma unroll
    for (int v = 0; v < 8; ++v) {
      const float val = acc[v] + bv;
      const size_t off = (size_t)(m0 + wm + mb + v) * N + col;
      if (OUT_HALF) ((_Float16*)Cout)[off] = (_Float16)val;
      else          ((float*)Cout)[off] = val;
    }
  }
}

// ---------------------------------------------------------------------------
// GRU recurrent scan. grid = 512 (seq*2dirs), block = 128 (one thread / h).
// xw: [NTOK][768] (dir0 gates 0..383 | dir1 384..767), gates r,z,n each 128.
// hout: [NTOK][256] (fwd 0..127 | bwd 128..255)
// ---------------------------------------------------------------------------
__global__ void gru_scan(const float* __restrict__ xw,
                         const float* __restrict__ whh,   // (2,2,384,128)
                         const float* __restrict__ bhh,   // (2,2,384)
                         float* __restrict__ hout, int layer) {
  const int dir = blockIdx.x & 1;
  const int seq = blockIdx.x >> 1;
  const int j = threadIdx.x;
  const float* W  = whh + (size_t)(layer * 2 + dir) * 384 * 128;
  const float* bh = bhh + (layer * 2 + dir) * 384;
  __shared__ float hs[128];
  hs[j] = 0.0f;
  const float brj = bh[j], bzj = bh[128 + j], bnj = bh[256 + j];
  float h = 0.0f;
  __syncthreads();
  const float* wr = W + (size_t)j * 128;
  const float* wz = W + (size_t)(128 + j) * 128;
  const float* wn = W + (size_t)(256 + j) * 128;
  for (int step = 0; step < 64; ++step) {
    const int t = dir ? (63 - step) : step;
    const size_t tok = (size_t)seq * 64 + t;
    const float* xwt = xw + tok * 768 + dir * 384;
    float ghr = brj, ghz = bzj, ghn = bnj;
#pragma unroll 8
    for (int i = 0; i < 128; ++i) {
      const float hv = hs[i];
      ghr = fmaf(wr[i], hv, ghr);
      ghz = fmaf(wz[i], hv, ghz);
      ghn = fmaf(wn[i], hv, ghn);
    }
    const float r = 1.0f / (1.0f + __expf(-(xwt[j] + ghr)));
    const float z = 1.0f / (1.0f + __expf(-(xwt[128 + j] + ghz)));
    const float nn = tanhf(xwt[256 + j] + r * ghn);
    h = (1.0f - z) * nn + z * h;
    __syncthreads();
    hs[j] = h;
    hout[tok * 256 + dir * 128 + j] = h;
    __syncthreads();
  }
}

// ---------------------------------------------------------------------------
// x1 = x + gru_out ; per-feature sum/sumsq (feature == tid since stride%256==0)
// ---------------------------------------------------------------------------
__global__ void init_stats(float* __restrict__ stats) {
  if (threadIdx.x < 512) stats[threadIdx.x] = 0.0f;
}

__global__ void resid_stats(const float* __restrict__ x,
                            const float* __restrict__ h2,
                            float* __restrict__ x1,
                            float* __restrict__ stats) {
  const int tid = threadIdx.x;
  float s = 0.0f, ss = 0.0f;
  const size_t total = (size_t)NTOK * FF;
  const size_t stride = (size_t)gridDim.x * 256;
  for (size_t i = (size_t)blockIdx.x * 256 + tid; i < total; i += stride) {
    const float v = x[i] + h2[i];
    x1[i] = v;
    s += v;
    ss += v * v;
  }
  atomicAdd(&stats[tid], s);
  atomicAdd(&stats[256 + tid], ss);
}

__global__ void bn_finalize(const float* __restrict__ stats,
                            const float* __restrict__ gamma,
                            const float* __restrict__ beta,
                            float* __restrict__ scsh) {
  const int c = threadIdx.x;
  const float n = (float)NTOK;
  const float mean = stats[c] / n;
  const float var = stats[256 + c] / n - mean * mean;
  const float inv = rsqrtf(var + EPSV);
  const float sc = gamma[c] * inv;
  scsh[c] = sc;
  scsh[256 + c] = beta[c] - mean * sc;
}

// ---------------------------------------------------------------------------
// Conv as implicit GEMM: M=NTOK rows=(b,s), N=256 out-ch, K=F*CK=1280.
// A[row][kk] = BN(x1[b, s+k-2, ci]) with kk = ci*5+k, zero-padded along s.
// x2 = x1 + leaky(conv + bias).
// ---------------------------------------------------------------------------
__global__ void conv_gemm(const float* __restrict__ x1,
                          const float* __restrict__ scsh,
                          const float* __restrict__ convw,  // [256][1280]
                          const float* __restrict__ convb,
                          float* __restrict__ x2) {
  __shared__ _Float16 As[64 * 32];
  __shared__ _Float16 Bs[64 * 32];
  const int tid  = threadIdx.x;
  const int wave = tid >> 5;
  const int m0 = blockIdx.x * 64;
  const int n0 = blockIdx.y * 64;
  const int wm = (wave & 3) * 16;
  const int wn = (wave >> 2) * 32;
  v8f acc0 = {}; v8f acc1 = {};

  for (int k0 = 0; k0 < FF * CKK; k0 += 32) {
    for (int i = tid; i < 64 * 32; i += 256) {
      const int r = i >> 5, c = i & 31;
      const int kk = k0 + c;
      const int ci = kk / CKK;
      const int kt = kk - ci * CKK;
      const int row = m0 + r;
      const int b = row >> 9;
      const int s = row & 511;
      const int sp = s + kt - (CKK / 2);
      float v = 0.0f;
      if (sp >= 0 && sp < SS)
        v = scsh[ci] * x1[((size_t)(b << 9) + sp) * FF + ci] + scsh[256 + ci];
      As[i] = (_Float16)v;
    }
    for (int i = tid; i < 64 * 32; i += 256) {
      const int r = i >> 5, c = i & 31;
      Bs[i] = (_Float16)convw[(size_t)(n0 + r) * (FF * CKK) + k0 + c];
    }
    __syncthreads();
    const v16h af  = ld_a_frag(&As[wm * 32], 32);
    const v16h bf0 = ld_b_frag(&Bs[wn * 32], 32);
    const v16h bf1 = ld_b_frag(&Bs[(wn + 16) * 32], 32);
    acc0 = wmma16(af, bf0, acc0);
    acc1 = wmma16(af, bf1, acc1);
    __syncthreads();
  }

  const int lane = tid & 31;
  const int n  = lane & 15;
  const int mb = (lane >> 4) * 8;
#pragma unroll
  for (int t = 0; t < 2; ++t) {
    const v8f acc = t ? acc1 : acc0;
    const int col = n0 + wn + t * 16 + n;
    const float bv = convb[col];
#pragma unroll
    for (int v = 0; v < 8; ++v) {
      float cv = acc[v] + bv;
      cv = cv >= 0.0f ? cv : SLOPE * cv;
      const size_t off = (size_t)(m0 + wm + mb + v) * FF + col;
      x2[off] = x1[off] + cv;
    }
  }
}

// ---------------------------------------------------------------------------
// Attention: per (b,h), flash-style. Each wave owns 16 q rows; D=32 so one
// WMMA (K=32) per 16x16 score tile. Online softmax via 16-lane shfl_xor
// (C-layout keeps a row inside one 16-lane half). P relayout via LDS.
// Epilogue: /l, leaky, LayerNorm over D, +x2 -> out.
// ---------------------------------------------------------------------------
__global__ void attention(const _Float16* __restrict__ Qh,
                          const _Float16* __restrict__ Kh,
                          const _Float16* __restrict__ Vh,
                          const float* __restrict__ x2,
                          const float* __restrict__ rel_emb,  // (8,512)
                          const float* __restrict__ ln_g,     // (8,32)
                          const float* __restrict__ ln_b,
                          float* __restrict__ out) {
  const int bh = blockIdx.x;
  const int b = bh >> 3, h = bh & 7;
  const int wave = threadIdx.x >> 5;
  const int lane = threadIdx.x & 31;
  const int q0 = blockIdx.y * 128 + wave * 16;
  const size_t tokq0 = (size_t)b * SS + q0;
  const int n_c = lane & 15;
  const int mb = (lane >> 4) * 8;

  __shared__ _Float16 ptile[8 * 16 * 32];
  _Float16* pt = ptile + wave * 16 * 32;

  // Q fragment (A-layout), loaded once
  v16h qf;
  {
    const int m = lane & 15;
    const int kh = (lane >> 4) << 4;
    const _Float16* p = Qh + (tokq0 + m) * FF + h * DD;
#pragma unroll
    for (int i = 0; i < 16; ++i) qf[i] = p[kh + i];
  }

  const float scale = 0.17677669529663687f;  // 1/sqrt(32)
  const float* rel = rel_emb + h * 512;
  float mrow[8], lrow[8];
#pragma unroll
  for (int v = 0; v < 8; ++v) { mrow[v] = -1e30f; lrow[v] = 0.0f; }
  v8f o0 = {}; v8f o1 = {};

  for (int t0 = 0; t0 < SS; t0 += 32) {
    // score tiles (cols t0..t0+15 and t0+16..t0+31)
    v16h kf0, kf1;
    {
      const int n = lane & 15;
      const int ko = (lane >> 4) << 4;
      const _Float16* p0 = Kh + ((size_t)b * SS + t0 + n) * FF + h * DD + ko;
      const _Float16* p1 = Kh + ((size_t)b * SS + t0 + 16 + n) * FF + h * DD + ko;
#pragma unroll
      for (int i = 0; i < 16; ++i) { kf0[i] = p0[i]; kf1[i] = p1[i]; }
    }
    v8f s0 = {}; v8f s1 = {};
    s0 = wmma16(qf, kf0, s0);
    s1 = wmma16(qf, kf1, s1);

    float sv0[8], sv1[8], nm[8];
#pragma unroll
    for (int v = 0; v < 8; ++v) {
      const int m = q0 + mb + v;
      int d0 = m - (t0 + n_c);        if (d0 < 0) d0 = -d0;
      int d1 = m - (t0 + 16 + n_c);   if (d1 < 0) d1 = -d1;
      sv0[v] = s0[v] * scale + rel[d0];
      sv1[v] = s1[v] * scale + rel[d1];
      nm[v] = fmaxf(sv0[v], sv1[v]);
    }
#pragma unroll
    for (int off = 1; off < 16; off <<= 1)
#pragma unroll
      for (int v = 0; v < 8; ++v) nm[v] = fmaxf(nm[v], __shfl_xor(nm[v], off, 32));

    float corr[8], rsum[8];
#pragma unroll
    for (int v = 0; v < 8; ++v) {
      const float newm = fmaxf(mrow[v], nm[v]);
      corr[v] = __expf(mrow[v] - newm);
      mrow[v] = newm;
      sv0[v] = __expf(sv0[v] - newm);
      sv1[v] = __expf(sv1[v] - newm);
      rsum[v] = sv0[v] + sv1[v];
    }
#pragma unroll
    for (int off = 1; off < 16; off <<= 1)
#pragma unroll
      for (int v = 0; v < 8; ++v) rsum[v] += __shfl_xor(rsum[v], off, 32);
#pragma unroll
    for (int v = 0; v < 8; ++v) {
      lrow[v] = lrow[v] * corr[v] + rsum[v];
      o0[v] *= corr[v];
      o1[v] *= corr[v];
    }

    // P (C-layout) -> LDS [16][32] row-major -> reload as A fragment
#pragma unroll
    for (int v = 0; v < 8; ++v) {
      pt[(mb + v) * 32 + n_c]      = (_Float16)sv0[v];
      pt[(mb + v) * 32 + 16 + n_c] = (_Float16)sv1[v];
    }
    __syncthreads();
    const v16h pf = ld_a_frag(pt, 32);

    // V fragments: B[k=t][n=d]
    v16h vf0, vf1;
    {
      const int n = lane & 15;
      const int ko = (lane >> 4) << 4;
      const _Float16* pv = Vh + ((size_t)b * SS + t0 + ko) * FF + h * DD;
#pragma unroll
      for (int i = 0; i < 16; ++i) {
        vf0[i] = pv[(size_t)i * FF + n];
        vf1[i] = pv[(size_t)i * FF + 16 + n];
      }
    }
    o0 = wmma16(pf, vf0, o0);
    o1 = wmma16(pf, vf1, o1);
    __syncthreads();
  }

  // epilogue: softmax normalize, leaky, LayerNorm over D=32, residual
  float a0[8], a1[8], sum[8];
#pragma unroll
  for (int v = 0; v < 8; ++v) {
    const float il = 1.0f / lrow[v];
    float x0 = o0[v] * il, x1v = o1[v] * il;
    x0 = x0 >= 0.0f ? x0 : SLOPE * x0;
    x1v = x1v >= 0.0f ? x1v : SLOPE * x1v;
    a0[v] = x0; a1[v] = x1v;
    sum[v] = x0 + x1v;
  }
#pragma unroll
  for (int off = 1; off < 16; off <<= 1)
#pragma unroll
    for (int v = 0; v < 8; ++v) sum[v] += __shfl_xor(sum[v], off, 32);
  float mu[8], sq[8];
#pragma unroll
  for (int v = 0; v < 8; ++v) {
    mu[v] = sum[v] * (1.0f / DD);
    const float d0 = a0[v] - mu[v], d1 = a1[v] - mu[v];
    sq[v] = d0 * d0 + d1 * d1;
  }
#pragma unroll
  for (int off = 1; off < 16; off <<= 1)
#pragma unroll
    for (int v = 0; v < 8; ++v) sq[v] += __shfl_xor(sq[v], off, 32);

  const float g0 = ln_g[h * DD + n_c],      b0 = ln_b[h * DD + n_c];
  const float g1 = ln_g[h * DD + 16 + n_c], b1 = ln_b[h * DD + 16 + n_c];
#pragma unroll
  for (int v = 0; v < 8; ++v) {
    const float inv = rsqrtf(sq[v] * (1.0f / DD) + EPSV);
    const size_t tok = tokq0 + mb + v;
    const size_t off0 = tok * FF + h * DD + n_c;
    const size_t off1 = off0 + 16;
    out[off0] = x2[off0] + (a0[v] - mu[v]) * inv * g0 + b0;
    out[off1] = x2[off1] + (a1[v] - mu[v]) * inv * g1 + b1;
  }
}

// ---------------------------------------------------------------------------
extern "C" void kernel_launch(void* const* d_in, const int* in_sizes, int n_in,
                              void* d_out, int out_size, void* d_ws, size_t ws_size,
                              hipStream_t stream) {
  (void)in_sizes; (void)n_in; (void)out_size; (void)ws_size;
  const float* x        = (const float*)d_in[0];
  const float* gru_w_ih = (const float*)d_in[1];   // (2,2,384,256)
  const float* gru_w_hh = (const float*)d_in[2];   // (2,2,384,128)
  const float* gru_b_ih = (const float*)d_in[3];   // (2,2,384)
  const float* gru_b_hh = (const float*)d_in[4];
  const float* bn_gamma = (const float*)d_in[5];
  const float* bn_beta  = (const float*)d_in[6];
  const float* conv_w   = (const float*)d_in[7];   // (256,256,5,1)
  const float* conv_b   = (const float*)d_in[8];
  const float* wq       = (const float*)d_in[9];   // (8,32,256)
  const float* wk       = (const float*)d_in[10];
  const float* wv       = (const float*)d_in[11];
  const float* rel_emb  = (const float*)d_in[12];  // (8,512)
  const float* ln_g     = (const float*)d_in[13];
  const float* ln_b     = (const float*)d_in[14];
  float* out = (float*)d_out;

  float* ws = (float*)d_ws;
  float* xw    = ws;                              // 16384*768
  float* h1    = xw + (size_t)NTOK * 768;         // 16384*256
  float* h2    = h1 + (size_t)NTOK * FF;
  float* x1    = h2 + (size_t)NTOK * FF;
  float* x2    = x1 + (size_t)NTOK * FF;
  float* stats = x2 + (size_t)NTOK * FF;          // 512
  float* scsh  = stats + 512;                     // 512
  _Float16* Qh = (_Float16*)(scsh + 512);         // 16384*256 halves
  _Float16* Kh = Qh + (size_t)NTOK * FF;
  _Float16* Vh = Kh + (size_t)NTOK * FF;

  const dim3 blk(256);

  hipLaunchKernelGGL(init_stats, dim3(1), dim3(512), 0, stream, stats);

  // ---- GRU layer 0: xw GEMM (both dirs as one 16384x768x256) + scan
  hipLaunchKernelGGL((wmma_gemm_nk<0>), dim3(NTOK / 64, 768 / 64), blk, 0, stream,
                     x, gru_w_ih, gru_b_ih, (void*)xw, NTOK, 768, FF);
  hipLaunchKernelGGL(gru_scan, dim3(512), dim3(128), 0, stream,
                     xw, gru_w_hh, gru_b_hh, h1, 0);

  // ---- GRU layer 1
  hipLaunchKernelGGL((wmma_gemm_nk<0>), dim3(NTOK / 64, 768 / 64), blk, 0, stream,
                     h1, gru_w_ih + (size_t)2 * 384 * 256, gru_b_ih + 768,
                     (void*)xw, NTOK, 768, FF);
  hipLaunchKernelGGL(gru_scan, dim3(512), dim3(128), 0, stream,
                     xw, gru_w_hh, gru_b_hh, h2, 1);

  // ---- residual + BN stats + finalize
  hipLaunchKernelGGL(resid_stats, dim3(1024), blk, 0, stream, x, h2, x1, stats);
  hipLaunchKernelGGL(bn_finalize, dim3(1), dim3(256), 0, stream,
                     stats, bn_gamma, bn_beta, scsh);

  // ---- conv (implicit GEMM K=1280) + leaky + residual -> x2
  hipLaunchKernelGGL(conv_gemm, dim3(NTOK / 64, FF / 64), blk, 0, stream,
                     x1, scsh, conv_w, conv_b, x2);

  // ---- QKV projections (emit f16 for the attention WMMAs)
  hipLaunchKernelGGL((wmma_gemm_nk<1>), dim3(NTOK / 64, FF / 64), blk, 0, stream,
                     x2, wq, (const float*)nullptr, (void*)Qh, NTOK, FF, FF);
  hipLaunchKernelGGL((wmma_gemm_nk<1>), dim3(NTOK / 64, FF / 64), blk, 0, stream,
                     x2, wk, (const float*)nullptr, (void*)Kh, NTOK, FF, FF);
  hipLaunchKernelGGL((wmma_gemm_nk<1>), dim3(NTOK / 64, FF / 64), blk, 0, stream,
                     x2, wv, (const float*)nullptr, (void*)Vh, NTOK, FF, FF);

  // ---- fused attention + leaky + LayerNorm + residual -> out
  hipLaunchKernelGGL(attention, dim3(BB * NHH, SS / 128), blk, 0, stream,
                     Qh, Kh, Vh, x2, rel_emb, ln_g, ln_b, out);
}
